// SelfAttn_78546361909708
// MI455X (gfx1250) — compile-verified, matrix operand order checked by reading, not run on hardware
//
#include <hip/hip_runtime.h>
#include <hip/hip_bf16.h>
#include <math.h>

// CDNA5 / gfx1250, wave32. Fused SAGAN self-attention.
//   Pass 1: 1x1-conv projections (theta, phi, g) + 2x2 maxpool, f32 -> f16 workspace.
//   Pass 2: flash-attention per 16-query tile per wave using v_wmma_f32_16x16x32_f16:
//           S^T = phi_tile(A) x theta(B); online softmax in f32; P packs in-lane
//           (D layout -> B layout is a no-shuffle identity for this orientation);
//           O^T += g_tile(A) x P(B); final w_o projection as a 4th WMMA; residual + gamma.

typedef __attribute__((ext_vector_type(16))) _Float16 v16h;
typedef __attribute__((ext_vector_type(8)))  float    v8f;
typedef __attribute__((ext_vector_type(8)))  _Float16 h8;

#define NB   16      // batch
#define NC   64      // channels
#define NHW  4096    // 64*64 pixels (queries)
#define NKV  1024    // pooled pixels (keys/values)

// ---------------------------------------------------------------------------
// Pass 1: projections + pooling. One thread per pooled (2x2) pixel.
//   theta_t : [B, 4096, 8]  f16  (query-major rows -> contiguous 16B per query)
//   phi_t   : [B, 1024, 8]  f16  (key-major rows)
//   g_v     : [B, 32, 1024] f16  (value-channel-major, contiguous in key)
//   wo16    : [64, 32]      f16
// ---------------------------------------------------------------------------
__global__ __launch_bounds__(256) void sagan_proj(
    const float* __restrict__ x,
    const float* __restrict__ w_theta,
    const float* __restrict__ w_phi,
    const float* __restrict__ w_g,
    const float* __restrict__ w_o,
    _Float16* __restrict__ theta_t,
    _Float16* __restrict__ phi_t,
    _Float16* __restrict__ g_v,
    _Float16* __restrict__ wo16)
{
    const int tid = blockIdx.x * blockDim.x + threadIdx.x;     // 0 .. 16383
    if (tid < 64 * 32) wo16[tid] = (_Float16)w_o[tid];

    const int b   = tid >> 10;
    const int hw2 = tid & 1023;
    const int h2  = hw2 >> 5;
    const int w2  = hw2 & 31;
    const float* xb = x + (size_t)b * NC * NHW;

    float pmax[8], gmax[32];
#pragma unroll
    for (int o = 0; o < 8;  ++o) pmax[o] = -INFINITY;
#pragma unroll
    for (int o = 0; o < 32; ++o) gmax[o] = -INFINITY;

#pragma unroll
    for (int py = 0; py < 2; ++py) {
#pragma unroll
        for (int px = 0; px < 2; ++px) {
            const int h   = 2 * h2 + py;
            const int w   = 2 * w2 + px;
            const int pix = h * 64 + w;
            const float* xp = xb + pix;

            float at[8] = {0}, ap[8] = {0}, ag[32] = {0};
            for (int c = 0; c < 64; ++c) {
                const float xv = xp[(size_t)c * NHW];
#pragma unroll
                for (int o = 0; o < 8;  ++o) at[o] += w_theta[o * 64 + c] * xv;
#pragma unroll
                for (int o = 0; o < 8;  ++o) ap[o] += w_phi[o * 64 + c] * xv;
#pragma unroll
                for (int o = 0; o < 32; ++o) ag[o] += w_g[o * 64 + c] * xv;
            }
            h8 trow;
#pragma unroll
            for (int o = 0; o < 8; ++o) trow[o] = (_Float16)at[o];
            *(h8*)(theta_t + ((size_t)b * NHW + pix) * 8) = trow;
#pragma unroll
            for (int o = 0; o < 8;  ++o) pmax[o] = fmaxf(pmax[o], ap[o]);
#pragma unroll
            for (int o = 0; o < 32; ++o) gmax[o] = fmaxf(gmax[o], ag[o]);
        }
    }
    h8 prow;
#pragma unroll
    for (int o = 0; o < 8; ++o) prow[o] = (_Float16)pmax[o];
    *(h8*)(phi_t + ((size_t)b * NKV + hw2) * 8) = prow;
#pragma unroll
    for (int o = 0; o < 32; ++o)
        g_v[((size_t)b * 32 + o) * NKV + hw2] = (_Float16)gmax[o];
}

// ---------------------------------------------------------------------------
// Pass 2: flash attention. One wave = one 16-query tile; 32 iterations of
// 32 keys each. All WMMA operands are built with contiguous b128 loads and
// in-lane f32->f16 packs (D layout of S^T == B layout of P for this mapping).
// ---------------------------------------------------------------------------
__global__ __launch_bounds__(256) void sagan_attn(
    const float* __restrict__ x,
    const float* __restrict__ gamma_p,
    const _Float16* __restrict__ theta_t,
    const _Float16* __restrict__ phi_t,
    const _Float16* __restrict__ g_v,
    const _Float16* __restrict__ wo16,
    float* __restrict__ out)
{
    const int lane = threadIdx.x & 31;
    const int half = lane >> 4;          // which 16-lane half of the wave
    const int ql   = lane & 15;
    const int wave = (blockIdx.x * blockDim.x + threadIdx.x) >> 5;
    const int b    = wave >> 8;          // 256 query tiles per batch
    const int qt   = wave & 255;
    const int n    = qt * 16 + ql;       // global query index

    // B operand: theta for this tile's 16 queries (dims 0..7, K padded to 32).
    // B layout: lanes 0-15 supply K=0..7 (+16..23), lanes 16-31 supply K=8..15
    // (+24..31) -> all zero padding for head dim 8, so only half0 loads.
    v16h bq = {};
    if (half == 0) {
        h8 q8 = *(const h8*)(theta_t + ((size_t)b * NHW + n) * 8);
#pragma unroll
        for (int i = 0; i < 8; ++i) bq[i] = q8[i];
    }

    float m = -INFINITY, s = 0.f;
    v8f o0 = {};   // O^T rows v = 0..15   (D layout: v = vgpr + 8*half, q = lane%16)
    v8f o1 = {};   // O^T rows v = 16..31
    const _Float16* gb = g_v + (size_t)b * 32 * NKV;
    const _Float16* pb = phi_t + (size_t)b * NKV * 8;

    for (int kt = 0; kt < 32; ++kt) {
        const int kb = kt * 32;

        // A operands: phi rows (keys) for two 16-key score tiles.
        v16h ap0 = {}, ap1 = {};
        if (half == 0) {
            h8 k0 = *(const h8*)(pb + (size_t)(kb + ql) * 8);
            h8 k1 = *(const h8*)(pb + (size_t)(kb + 16 + ql) * 8);
#pragma unroll
            for (int i = 0; i < 8; ++i) { ap0[i] = k0[i]; ap1[i] = k1[i]; }
        }
        v8f zc = {};
        // S^T tiles: D[M=key, N=query]
        v8f d0 = __builtin_amdgcn_wmma_f32_16x16x32_f16(false, ap0, false, bq, (short)0, zc, false, false);
        v8f d1 = __builtin_amdgcn_wmma_f32_16x16x32_f16(false, ap1, false, bq, (short)0, zc, false, false);

        // Online softmax over this lane's 16 key scores (its query column),
        // merged with the other lane half (which owns the other 16 keys... of
        // each tile: keys r+8; combined via xor-16 shuffle).
        float tmax = d0[0];
#pragma unroll
        for (int r = 1; r < 8; ++r) tmax = fmaxf(tmax, d0[r]);
#pragma unroll
        for (int r = 0; r < 8; ++r) tmax = fmaxf(tmax, d1[r]);
        tmax = fmaxf(tmax, __shfl_xor(tmax, 16, 32));
        const float mnew  = fmaxf(m, tmax);
        const float scale = __expf(m - mnew);     // exp(-inf)=0 on first tile
        s *= scale;
#pragma unroll
        for (int r = 0; r < 8; ++r) { o0[r] *= scale; o1[r] *= scale; }

        // P in B-matrix layout: in-lane pack. Lanes 0-15 own K=0..7 & 16..23,
        // lanes 16-31 own K=8..15 & 24..31 -- exactly the keys this lane holds.
        v16h bp;
#pragma unroll
        for (int r = 0; r < 8; ++r) {
            const float p0 = __expf(d0[r] - mnew);
            const float p1 = __expf(d1[r] - mnew);
            s += p0 + p1;
            bp[r]     = (_Float16)p0;
            bp[8 + r] = (_Float16)p1;
        }
        m = mnew;

        // A operands: g value rows, 32 keys of this iteration (full K).
        v16h a0, a1;
        {
            const _Float16* g0 = gb + (size_t)ql * NKV + kb;
            const _Float16* g1 = gb + (size_t)(16 + ql) * NKV + kb;
            h8 t0a = *(const h8*)(g0 + half * 8);        // K=0..7  / 8..15
            h8 t0b = *(const h8*)(g0 + 16 + half * 8);   // K=16..23 / 24..31
            h8 t1a = *(const h8*)(g1 + half * 8);
            h8 t1b = *(const h8*)(g1 + 16 + half * 8);
#pragma unroll
            for (int i = 0; i < 8; ++i) {
                a0[i] = t0a[i]; a0[8 + i] = t0b[i];
                a1[i] = t1a[i]; a1[8 + i] = t1b[i];
            }
        }
        o0 = __builtin_amdgcn_wmma_f32_16x16x32_f16(false, a0, false, bp, (short)0, o0, false, false);
        o1 = __builtin_amdgcn_wmma_f32_16x16x32_f16(false, a1, false, bp, (short)0, o1, false, false);
    }

    // Finish softmax normalization (merge lane-half partial sums).
    const float stot = s + __shfl_xor(s, 16, 32);
    const float inv  = 1.f / stot;

    // O^T (D layout) -> B-matrix layout for the w_o projection: in-lane pack.
    v16h bo;
#pragma unroll
    for (int r = 0; r < 8; ++r) {
        bo[r]     = (_Float16)(o0[r] * inv);   // K = v 0..7 (half0) / 8..15 (half1)
        bo[8 + r] = (_Float16)(o1[r] * inv);   // K = v 16..23      / 24..31
    }

    const float gamma = *gamma_p;
    const float* xb = x   + (size_t)b * NC * NHW;
    float*       ob = out + (size_t)b * NC * NHW;

#pragma unroll
    for (int ct = 0; ct < 4; ++ct) {           // 4 tiles of 16 output channels
        v16h aw;
        const _Float16* wrow = wo16 + (size_t)(ct * 16 + ql) * 32;
        h8 wlo = *(const h8*)(wrow + half * 8);
        h8 whi = *(const h8*)(wrow + 16 + half * 8);
#pragma unroll
        for (int i = 0; i < 8; ++i) { aw[i] = wlo[i]; aw[8 + i] = whi[i]; }
        v8f zc = {};
        v8f dd = __builtin_amdgcn_wmma_f32_16x16x32_f16(false, aw, false, bo, (short)0, zc, false, false);
#pragma unroll
        for (int r = 0; r < 8; ++r) {
            const int    co  = ct * 16 + half * 8 + r;
            const size_t idx = (size_t)co * NHW + n;
            ob[idx] = gamma * dd[r] + xb[idx];
        }
    }
}

// ---------------------------------------------------------------------------
extern "C" void kernel_launch(void* const* d_in, const int* in_sizes, int n_in,
                              void* d_out, int out_size, void* d_ws, size_t ws_size,
                              hipStream_t stream) {
    const float* x       = (const float*)d_in[0];
    const float* w_theta = (const float*)d_in[1];
    const float* w_phi   = (const float*)d_in[2];
    const float* w_g     = (const float*)d_in[3];
    const float* w_o     = (const float*)d_in[4];
    const float* gamma   = (const float*)d_in[5];
    float* out = (float*)d_out;

    // Workspace layout (f16), ~2.36 MB total:
    char* ws = (char*)d_ws;
    _Float16* theta_t = (_Float16*)(ws);                        // 16*4096*8  = 1 MB
    _Float16* phi_t   = (_Float16*)(ws + (1u << 20));           // 16*1024*8  = 256 KB
    _Float16* g_v     = (_Float16*)(ws + (1u << 20) + (1u << 18)); // 16*32*1024 = 1 MB
    _Float16* wo16    = (_Float16*)(ws + (2u << 20) + (1u << 18) + (1u << 16));

    // Pass 1: one thread per pooled pixel, 16 * 1024 threads.
    sagan_proj<<<dim3(64), dim3(256), 0, stream>>>(
        x, w_theta, w_phi, w_g, w_o, theta_t, phi_t, g_v, wo16);

    // Pass 2: one wave per 16-query tile: 16 batches * 256 tiles = 4096 waves.
    sagan_attn<<<dim3(512), dim3(256), 0, stream>>>(
        x, gamma, theta_t, phi_t, g_v, wo16, out);
}
